// MeshDecoder_66030827208810
// MI455X (gfx1250) — matrix operand order repaired and checked by reading, not compiled, not run on hardware
//
#include <hip/hip_runtime.h>
#include <hip/hip_bf16.h>

// ---------------------------------------------------------------------------
// MeshDecoder on MI455X (gfx1250, wave32).
//
// Exact simplification: the reference's min/max over identical batch copies
// makes the normalized grid coordinate constant (-2/3 per axis), so trilinear
// sampling collapses to B*C scalars and encoder layer 1 to
// vertices @ W1[:3,:] + per-batch constant.
//
// 232 GFLOP GEMM chain -> v_wmma_f32_16x16x32_bf16 (f32 accumulate).
// All GEMM operands are stored bf16 in global (identical rounding to doing
// the fp32->bf16 conversion at tile-staging time), so tiles are copied
// global->LDS with the Tensor Data Mover (tensor_load_to_lds, TENSORcnt),
// with D# LDS padding reproducing the bank-conflict-free 80B row stride.
// GCN edge aggregation: global_atomic_add_f32 (bandwidth bound).
// ---------------------------------------------------------------------------

typedef __attribute__((ext_vector_type(16))) __bf16 v16bf;
typedef __attribute__((ext_vector_type(8)))  __bf16 v8bf;
typedef __attribute__((ext_vector_type(8)))  float  v8f;

#define TM 128
#define TN 128
#define TK 32
#define LDT (TK + 8)   // padded LDS row stride: 40 elems = 80B (16B multiple)

#if defined(__gfx1250__) && __has_builtin(__builtin_amdgcn_tensor_load_to_lds)
#define USE_TDM 1
#else
#define USE_TDM 0
#endif

#if USE_TDM
typedef __attribute__((ext_vector_type(4))) unsigned int u32x4;
typedef __attribute__((ext_vector_type(8))) int          i32x8;
typedef __attribute__((ext_vector_type(4))) int          i32x4;

// DMA a 16-row x 32-col bf16 tile (rows of 64B) global -> LDS, padding LDS
// by 16B after each row (row stride 80B == LDT*2). rowsValid clamps OOB rows
// (TDM returns zeros beyond tensor_dim1).  ISA 08_async_tensor §8.
__device__ __forceinline__ void tdm_tile16(const __bf16* g, unsigned ldsOff,
                                           int rowsValid, unsigned rowLen)
{
    unsigned long long ga = (unsigned long long)(size_t)g;
    unsigned rl = (rowsValid < 0) ? 0u : (unsigned)rowsValid;
    u32x4 g0 = { 1u,                                    // count=1, user D#
                 ldsOff,                                // lds_addr
                 (unsigned)(ga & 0xFFFFFFFFull),        // global_addr[31:0]
                 (unsigned)((ga >> 32) & 0x1FFFFFFull)  // global_addr[56:32]
                 | (2u << 30) };                        // type = 2 ("image")
    i32x8 g1 = {
        (int)((1u << 16)      // data_size = 2 bytes
            | (1u << 20)      // pad_enable
            | (3u << 22)      // pad_interval: 2^(3+1) = 16 DWORDs (64B)
            | (3u << 25)),    // pad_amount: 4 DWORDs (16B)
        (int)((rowLen & 0xFFFFu) << 16),                    // tensor_dim0 lo
        (int)((rowLen >> 16) | ((rl & 0xFFFFu) << 16)),     // dim0 hi | dim1 lo
        (int)((rl >> 16) | (32u << 16)),                    // dim1 hi | tile_dim0=32
        (int)16,                                            // tile_dim1 = 16 rows
        (int)rowLen,                                        // tensor_dim0_stride lo
        0, 0 };                                             // stride hi, dim1_stride
    i32x4 gz = { 0, 0, 0, 0 };
#if defined(__clang_major__) && (__clang_major__ >= 23)
    // amdgpu-toolchain / therock-10.0 headers: 6-arg form (extra int32x8).
    i32x8 gz8 = { 0, 0, 0, 0, 0, 0, 0, 0 };
    __builtin_amdgcn_tensor_load_to_lds(g0, g1, gz, gz, gz8, 0);
#else
    // ROCm 7.2 (clang-22): 5-arg form.
    __builtin_amdgcn_tensor_load_to_lds(g0, g1, gz, gz, 0);
#endif
}
#endif

// ---------------------------------------------------------------------------
// WMMA GEMM:  C[M,N] = epilogue( A[M,K] (bf16) @ Wt[N,K] (bf16, k-fastest) )
// 256 threads = 8 waves in a 2(M) x 4(N) grid; each wave computes 64x32 via
// 4x2 accumulators of v_wmma_f32_16x16x32_bf16.
// M % 128 == 0 (M = 80000) and K % 32 == 0 hold for all uses here.
// flags: bit0 = relu, bit1 = write bf16 (else fp32).
// ---------------------------------------------------------------------------
__global__ __launch_bounds__(256)
void k_gemm_bf16(const __bf16* __restrict__ A, const __bf16* __restrict__ Wt,
                 const float* __restrict__ bias, void* __restrict__ Cp,
                 int K, int N, int flags)
{
    __shared__ __align__(16) __bf16 As[TM][LDT];   // [m][k]
    __shared__ __align__(16) __bf16 Bs[TN][LDT];   // [n][k]

    const int tid   = threadIdx.x;
    const int wave  = tid >> 5;
    const int lane  = tid & 31;
    const int lhalf = lane >> 4;    // 0: lanes 0-15, 1: lanes 16-31
    const int lrow  = lane & 15;
    const int m0 = blockIdx.y * TM;
    const int n0 = blockIdx.x * TN;
    const int wm = (wave & 1) * 64;
    const int wn = (wave >> 1) * 32;

    v8f acc[4][2] = {};

    for (int k0 = 0; k0 < K; k0 += TK) {
#if USE_TDM
        {
            // Each wave DMAs its own 16-row slice of the A and B tiles and
            // waits on its own TENSORcnt; the barrier publishes all slices.
            unsigned ldsA = (unsigned)(size_t)(&As[0][0]) + (unsigned)(wave * 16 * LDT * 2);
            unsigned ldsB = (unsigned)(size_t)(&Bs[0][0]) + (unsigned)(wave * 16 * LDT * 2);
            tdm_tile16(A  + (size_t)(m0 + wave * 16) * K + k0, ldsA, 16, (unsigned)K);
            tdm_tile16(Wt + (size_t)(n0 + wave * 16) * K + k0, ldsB,
                       N - (n0 + wave * 16), (unsigned)K);
            __builtin_amdgcn_s_wait_tensorcnt((short)0);
        }
#else
        // Vectorized fallback: 8-element (16B) chunk copies.
        #pragma unroll
        for (int ch = tid; ch < (TM * TK) / 8; ch += 256) {
            int r = ch >> 2;              // 4 chunks per 32-elem row
            int c = (ch & 3) * 8;
            *(v8bf*)&As[r][c] = *(const v8bf*)&A[(size_t)(m0 + r) * K + k0 + c];
        }
        #pragma unroll
        for (int ch = tid; ch < (TN * TK) / 8; ch += 256) {
            int r = ch >> 2;
            int c = (ch & 3) * 8;
            v8bf v = {};
            if (n0 + r < N) v = *(const v8bf*)&Wt[(size_t)(n0 + r) * K + k0 + c];
            *(v8bf*)&Bs[r][c] = v;
        }
#endif
        __syncthreads();

        // B fragments (32x16, K x N): lane holds column N = lane%16,
        // elements 0..7 <-> K = (lane/16)*8 + 0..7, 8..15 <-> K = 16 + (lane/16)*8 + 0..7.
        v16bf bfrag[2];
        #pragma unroll
        for (int nt = 0; nt < 2; ++nt) {
            const __bf16* bp = &Bs[wn + nt * 16 + lrow][0];
            v8bf blo = *(const v8bf*)(bp + lhalf * 8);
            v8bf bhi = *(const v8bf*)(bp + 16 + lhalf * 8);
            bfrag[nt] = __builtin_shufflevector(blo, bhi,
                0, 1, 2, 3, 4, 5, 6, 7, 8, 9, 10, 11, 12, 13, 14, 15);
        }

        #pragma unroll
        for (int mt = 0; mt < 4; ++mt) {
            // A fragment (16x32, M x K): lane holds row M = lane%16;
            // elements 0..7  <-> K = (lane/16)*8 + 0..7
            // elements 8..15 <-> K = 16 + (lane/16)*8 + 0..7
            const __bf16* ap = &As[wm + mt * 16 + lrow][0];
            v8bf alo = *(const v8bf*)(ap + lhalf * 8);
            v8bf ahi = *(const v8bf*)(ap + 16 + lhalf * 8);
            v16bf afrag = __builtin_shufflevector(alo, ahi,
                0, 1, 2, 3, 4, 5, 6, 7, 8, 9, 10, 11, 12, 13, 14, 15);
            #pragma unroll
            for (int nt = 0; nt < 2; ++nt) {
                acc[mt][nt] = __builtin_amdgcn_wmma_f32_16x16x32_bf16(
                    false, afrag, false, bfrag[nt], (short)0, acc[mt][nt],
                    false, false);
            }
        }
        __syncthreads();
    }

    // Epilogue. C/D layout: n = lane%16, m = vgpr + 8*(lane/16).
    const int do_relu = flags & 1;
    const int out_bf  = flags & 2;
    #pragma unroll
    for (int mt = 0; mt < 4; ++mt) {
        #pragma unroll
        for (int nt = 0; nt < 2; ++nt) {
            int n = n0 + wn + nt * 16 + lrow;
            if (n >= N) continue;
            float bv = bias ? bias[n] : 0.0f;
            #pragma unroll
            for (int v = 0; v < 8; ++v) {
                int m = m0 + wm + mt * 16 + lhalf * 8 + v;
                float x = acc[mt][nt][v] + bv;
                if (do_relu) x = fmaxf(x, 0.0f);
                size_t idx = (size_t)m * N + n;
                if (out_bf) ((__bf16*)Cp)[idx] = (__bf16)x;
                else        ((float*) Cp)[idx] = x;
            }
        }
    }
}

// ---------------------------------------------------------------------------
// Small helper kernels
// ---------------------------------------------------------------------------

// Convert fp32 weight [K][N] -> bf16 transposed [N][K] (k-fastest).
__global__ void k_w_to_bf16_t(const float* __restrict__ in, __bf16* __restrict__ out,
                              int K, int N) {
    int i = blockIdx.x * blockDim.x + threadIdx.x;
    if (i >= K * N) return;
    int k = i / N, n = i % N;                 // coalesced read
    out[(size_t)n * K + k] = (__bf16)in[i];
}

__global__ void k_fill(float* __restrict__ p, float v, int n) {
    int i = blockIdx.x * blockDim.x + threadIdx.x;
    if (i < n) p[i] = v;
}

__global__ void k_deg(const int* __restrict__ dst, float* __restrict__ deg, int E) {
    int i = blockIdx.x * blockDim.x + threadIdx.x;
    if (i < E) atomicAdd(&deg[dst[i]], 1.0f);
}

__global__ void k_rsqrt_inplace(float* __restrict__ p, int n) {
    int i = blockIdx.x * blockDim.x + threadIdx.x;
    if (i < n) p[i] = rsqrtf(p[i]);
}

__global__ void k_edgenorm(const int* __restrict__ src, const int* __restrict__ dst,
                           const float* __restrict__ dinv, float* __restrict__ nrm, int E) {
    int i = blockIdx.x * blockDim.x + threadIdx.x;
    if (i < E) nrm[i] = dinv[src[i]] * dinv[dst[i]];
}

// Constant-coordinate trilinear sample (reference's grid collapses to -2/3).
__global__ void k_sample(const float* __restrict__ feat, float* __restrict__ vfeat,
                         int B, int C) {
    int i = blockIdx.x * blockDim.x + threadIdx.x;
    if (i >= B * C) return;
    const int S = 64;
    float vn = (2.0f * 0.0f / (0.0f + 1e-6f) - 1.0f) / 1.5f;   // = -2/3
    float x  = fminf(fmaxf((vn + 1.0f) * 0.5f * (float)(S - 1), 0.0f), (float)(S - 1));
    int   i0 = (int)floorf(x);
    float w  = x - (float)i0;
    int   i1 = min(i0 + 1, S - 1);
    const float* f = feat + (size_t)i * S * S * S;   // i = b*C + c
    #define G(z, y, xx) f[((z) * S + (y)) * S + (xx)]
    float c00 = G(i0, i0, i0) * (1.0f - w) + G(i0, i0, i1) * w;
    float c01 = G(i0, i1, i0) * (1.0f - w) + G(i0, i1, i1) * w;
    float c10 = G(i1, i0, i0) * (1.0f - w) + G(i1, i0, i1) * w;
    float c11 = G(i1, i1, i0) * (1.0f - w) + G(i1, i1, i1) * w;
    #undef G
    float c0 = c00 * (1.0f - w) + c01 * w;
    float c1 = c10 * (1.0f - w) + c11 * w;
    vfeat[i] = c0 * (1.0f - w) + c1 * w;
}

// const_b[b,o] = sum_c vfeat[b,c] * enc_W1[3+c, o] + enc_b1[o]   (o < 256)
__global__ void k_constb(const float* __restrict__ vfeat, const float* __restrict__ W1,
                         const float* __restrict__ b1, float* __restrict__ constb, int C) {
    int b = blockIdx.x, o = threadIdx.x;
    float s = b1[o];
    for (int c = 0; c < C; ++c)
        s = fmaf(vfeat[b * C + c], W1[(3 + c) * 256 + o], s);
    constb[b * 256 + o] = s;
}

// Encoder layer 1 (K=3, too thin for WMMA): relu(v @ W1[:3] + const_b) -> bf16
__global__ void k_enc1(const float* __restrict__ verts, const float* __restrict__ W1,
                       const float* __restrict__ constb, __bf16* __restrict__ out,
                       int N, int B) {
    long i = (long)blockIdx.x * blockDim.x + threadIdx.x;
    if (i >= (long)B * N * 256) return;
    int  o  = (int)(i & 255);
    long rn = i >> 8;                 // b*N + n
    int  n  = (int)(rn % N);
    int  b  = (int)(rn / N);
    float s = constb[b * 256 + o];
    s = fmaf(verts[n * 3 + 0], W1[o], s);
    s = fmaf(verts[n * 3 + 1], W1[256 + o], s);
    s = fmaf(verts[n * 3 + 2], W1[512 + o], s);
    out[i] = (__bf16)fmaxf(s, 0.0f);
}

// agg <- bias (broadcast over rows)
__global__ void k_agg_init(float* __restrict__ agg, const float* __restrict__ bias,
                           long total, int Cout) {
    long stride = (long)gridDim.x * blockDim.x;
    for (long i = (long)blockIdx.x * blockDim.x + threadIdx.x; i < total; i += stride)
        agg[i] = bias[(int)(i % Cout)];
}

// agg[b,dst] += T[b,src] * nrm[e]   (one wave per (edge, batch))
__global__ void k_scatter(const float* __restrict__ T, float* __restrict__ agg,
                          const int* __restrict__ src, const int* __restrict__ dst,
                          const float* __restrict__ nrm, int E, int N, int Cout, int B) {
    int  lane = threadIdx.x & 31;
    long gw = ((long)blockIdx.x * blockDim.x + threadIdx.x) >> 5;
    long nw = ((long)gridDim.x * blockDim.x) >> 5;
    long total = (long)E * B;
    for (long w = gw; w < total; w += nw) {
        int e = (int)(w % E);
        int b = (int)(w / E);
        float nv = nrm[e];
        const float* tp = T   + ((size_t)b * N + src[e]) * Cout;
        float*       ap = agg + ((size_t)b * N + dst[e]) * Cout;
        for (int c = lane; c < Cout; c += 32)
            atomicAdd(&ap[c], tp[c] * nv);
    }
}

// H = relu(T * dinv[n]^2 + agg) -> bf16   (agg holds bias + neighbor sum)
__global__ void k_combine(const float* __restrict__ T, const float* __restrict__ agg,
                          const float* __restrict__ dinv, __bf16* __restrict__ H,
                          int N, int Cout, long total) {
    long stride = (long)gridDim.x * blockDim.x;
    for (long i = (long)blockIdx.x * blockDim.x + threadIdx.x; i < total; i += stride) {
        long row = i / Cout;
        int  n   = (int)(row % N);
        float d  = dinv[n];
        H[i] = (__bf16)fmaxf(fmaf(T[i], d * d, agg[i]), 0.0f);
    }
}

// Final head: out = vertices + clip(nan_to_num(tanh(d2 @ hW3 + hb3)), -2.5, 2.5)
__global__ void k_head(const __bf16* __restrict__ D2, const float* __restrict__ W3,
                       const float* __restrict__ b3, const float* __restrict__ verts,
                       float* __restrict__ out, int N, int B) {
    long i = (long)blockIdx.x * blockDim.x + threadIdx.x;   // b*N + n
    if (i >= (long)B * N) return;
    int n = (int)(i % N);
    const __bf16* d = D2 + (size_t)i * 64;
    float s0 = b3[0], s1 = b3[1], s2 = b3[2];
    #pragma unroll 8
    for (int k = 0; k < 64; ++k) {
        float dv = (float)d[k];
        s0 = fmaf(dv, W3[k * 3 + 0], s0);
        s1 = fmaf(dv, W3[k * 3 + 1], s1);
        s2 = fmaf(dv, W3[k * 3 + 2], s2);
    }
    float s[3] = {s0, s1, s2};
    #pragma unroll
    for (int j = 0; j < 3; ++j) {
        float t = tanhf(s[j]);
        if (t != t) t = 0.0f;                       // nan_to_num
        t = fminf(fmaxf(t, -2.5f), 2.5f);           // clip
        out[(size_t)i * 3 + j] = verts[n * 3 + j] + t;
    }
}

// ---------------------------------------------------------------------------
extern "C" void kernel_launch(void* const* d_in, const int* in_sizes, int n_in,
                              void* d_out, int out_size, void* d_ws, size_t ws_size,
                              hipStream_t stream)
{
    const float* features = (const float*)d_in[0];
    const float* vertices = (const float*)d_in[1];
    const int*   edges    = (const int*)d_in[2];
    const float* enc_W1 = (const float*)d_in[3];
    const float* enc_b1 = (const float*)d_in[4];
    const float* enc_W2 = (const float*)d_in[5];
    const float* enc_b2 = (const float*)d_in[6];
    const float* gW[6]; const float* gb[6];
    for (int i = 0; i < 6; ++i) { gW[i] = (const float*)d_in[7 + 2 * i];
                                  gb[i] = (const float*)d_in[8 + 2 * i]; }
    const float* hW1 = (const float*)d_in[19]; const float* hb1 = (const float*)d_in[20];
    const float* hW2 = (const float*)d_in[21]; const float* hb2 = (const float*)d_in[22];
    const float* hW3 = (const float*)d_in[23]; const float* hb3 = (const float*)d_in[24];

    const int  C = 64;
    const int  N = in_sizes[1] / 3;                    // 20000
    const int  B = in_sizes[0] / (C * 64 * 64 * 64);   // 4
    const int  E = in_sizes[2] / 2;
    const int* esrc = edges;
    const int* edst = edges + E;
    const long ROWS = (long)B * N;                     // 80000 = 625 * 128

    // --- workspace layout --------------------------------------------------
    char* ws = (char*)d_ws;
    size_t off = 0;
    auto walloc = [&](size_t bytes) -> void* {
        void* p = ws + off;
        off = (off + bytes + 255) & ~(size_t)255;
        return p;
    };
    struct WInfo { const float* src; int k, n; __bf16* dst; };
    WInfo wi[9] = {
        {enc_W2, 256,  128, nullptr}, {gW[0], 128,  256, nullptr},
        {gW[1],  256,  512, nullptr}, {gW[2], 512, 1024, nullptr},
        {gW[3], 1024,  512, nullptr}, {gW[4], 512,  256, nullptr},
        {gW[5],  256,  128, nullptr}, {hW1,   128,  128, nullptr},
        {hW2,    128,   64, nullptr}
    };
    for (int i = 0; i < 9; ++i)
        wi[i].dst = (__bf16*)walloc((size_t)wi[i].k * wi[i].n * sizeof(__bf16));
    float*  deg    = (float*)walloc((size_t)N * 4);
    float*  nrm    = (float*)walloc((size_t)E * 4);
    float*  vfeat  = (float*)walloc((size_t)B * C * 4);
    float*  constb = (float*)walloc((size_t)B * 256 * 4);
    float*  bufT   = (float*)walloc((size_t)ROWS * 1024 * sizeof(float));   // GEMM out (fp32)
    float*  bufG   = (float*)walloc((size_t)ROWS * 1024 * sizeof(float));   // aggregation
    __bf16* bufH0  = (__bf16*)walloc((size_t)ROWS * 1024 * sizeof(__bf16)); // act ping
    __bf16* bufH1  = (__bf16*)walloc((size_t)ROWS * 1024 * sizeof(__bf16)); // act pong
    (void)ws_size; (void)n_in; (void)out_size;

    // --- precompute: bf16 transposed weights, graph norm, constant sample --
    for (int i = 0; i < 9; ++i) {
        int n = wi[i].k * wi[i].n;
        k_w_to_bf16_t<<<(n + 255) / 256, 256, 0, stream>>>(wi[i].src, wi[i].dst,
                                                           wi[i].k, wi[i].n);
    }
    k_fill<<<(N + 255) / 256, 256, 0, stream>>>(deg, 1.0f, N);     // self-loop
    k_deg<<<(E + 255) / 256, 256, 0, stream>>>(edst, deg, E);
    k_rsqrt_inplace<<<(N + 255) / 256, 256, 0, stream>>>(deg, N);  // deg -> dinv
    k_edgenorm<<<(E + 255) / 256, 256, 0, stream>>>(esrc, edst, deg, nrm, E);
    k_sample<<<(B * C + 255) / 256, 256, 0, stream>>>(features, vfeat, B, C);
    k_constb<<<B, 256, 0, stream>>>(vfeat, enc_W1, enc_b1, constb, C);

    // --- encoder -----------------------------------------------------------
    __bf16* Hcur = bufH0;
    __bf16* Hnxt = bufH1;
    {
        long tot = ROWS * 256;
        k_enc1<<<(int)((tot + 255) / 256), 256, 0, stream>>>(vertices, enc_W1,
                                                             constb, Hcur, N, B);
    }
    auto gemm = [&](const __bf16* Ain, const __bf16* Wb, const float* bias,
                    void* Cout_, int K, int Nn, int flags) {
        dim3 grid((Nn + TN - 1) / TN, (unsigned)(ROWS / TM));
        k_gemm_bf16<<<grid, 256, 0, stream>>>(Ain, Wb, bias, Cout_, K, Nn, flags);
    };
    auto swapH = [&]() { __bf16* t = Hcur; Hcur = Hnxt; Hnxt = t; };

    gemm(Hcur, wi[0].dst, enc_b2, Hnxt, 256, 128, 1 | 2);   // relu, bf16 out
    swapH();                                                // H [ROWS,128]

    // --- 6 GCN layers ------------------------------------------------------
    const int couts[6] = {256, 512, 1024, 512, 256, 128};
    int cin = 128;
    for (int l = 0; l < 6; ++l) {
        int co = couts[l];
        gemm(Hcur, wi[1 + l].dst, nullptr, bufT, cin, co, 0);   // T fp32
        long tot = ROWS * co;
        k_agg_init<<<4096, 256, 0, stream>>>(bufG, gb[l], tot, co);
        long waves = (long)E * B;
        long sb = (waves * 32 + 255) / 256;
        int  sblocks = (sb > 60000) ? 60000 : (int)sb;
        k_scatter<<<sblocks, 256, 0, stream>>>(bufT, bufG, esrc, edst, nrm,
                                               E, N, co, B);
        k_combine<<<4096, 256, 0, stream>>>(bufT, bufG, deg, Hnxt, N, co, tot);
        swapH();
        cin = co;
    }

    // --- head ---------------------------------------------------------------
    gemm(Hcur, wi[7].dst, hb1, Hnxt, 128, 128, 1 | 2); swapH();  // [ROWS,128]
    gemm(Hcur, wi[8].dst, hb2, Hnxt, 128,  64, 1 | 2); swapH();  // [ROWS,64]
    k_head<<<(int)((ROWS + 255) / 256), 256, 0, stream>>>(Hcur, hW3, hb3,
                                                          vertices,
                                                          (float*)d_out, N, B);
}
